// PointSpatioTemporalCorrelation_73735998538273
// MI455X (gfx1250) — matrix operand order repaired and checked by reading, not compiled, not run on hardware
//
#include <hip/hip_runtime.h>
#include <hip/hip_bf16.h>

typedef __attribute__((ext_vector_type(16))) _Float16 v16h;
typedef __attribute__((ext_vector_type(8)))  _Float16 v8h;
typedef __attribute__((ext_vector_type(8)))  float    v8f;

#define BQ_B   8
#define BQ_N   4096
#define BQ_C   64
#define BQ_OUT 64
#define BQ_NS  16
#define R2     0.01f      // RADIUS^2
#define KPAD   160        // 131 padded to 5*32
#define WAVES  8          // waves per block in fused kernel
#define WP_ELEMS (4 * 5 * 32 * 16)   // packed-W f16 element count (10240)

// ---------------------------------------------------------------------------
// Pack W (OUT=64 x F=131, f32) into this file's contiguous-K WMMA B-operand order, f16.
//   lane l -> n = ntile*16 + (l&15);  element e -> k = kstep*32 + (l>>4)*16 + e
// ---------------------------------------------------------------------------
__global__ __launch_bounds__(256) void pack_w(const float* __restrict__ W,
                                              _Float16* __restrict__ Wp) {
  int t = blockIdx.x * 256 + threadIdx.x;
  if (t >= WP_ELEMS) return;
  int e    = t & 15;
  int lane = (t >> 4) & 31;
  int ks   = (t >> 9) % 5;
  int nt   = t / (32 * 16 * 5);
  int k = ks * 32 + ((lane >> 4) << 4) + e;
  int n = nt * 16 + (lane & 15);
  float v = (k < 2 * BQ_C + 3) ? W[n * (2 * BQ_C + 3) + k] : 0.0f;
  Wp[t] = (_Float16)v;
}

// ---------------------------------------------------------------------------
// (B, C, N) -> (B, N, C) LDS-tiled transpose, coalesced on both sides.
// Grid: (N/64, B); block 256.  Tile 64(c) x 64(n), padded row vs bank conflicts.
// ---------------------------------------------------------------------------
__global__ __launch_bounds__(256) void transpose_cn(const float* __restrict__ src,
                                                    float* __restrict__ dst) {
  __shared__ float tile[64][65];
  const int b  = blockIdx.y;
  const int n0 = blockIdx.x * 64;
  const int tx = threadIdx.x & 63;   // fast index (n on load, c on store)
  const int ty = threadIdx.x >> 6;   // 0..3
#pragma unroll
  for (int i = 0; i < 64; i += 4) {
    int c = ty + i;
    tile[c][tx] = src[((size_t)b * BQ_C + c) * BQ_N + n0 + tx];  // coalesced over n
  }
  __syncthreads();
#pragma unroll
  for (int i = 0; i < 64; i += 4) {
    int nl = ty + i;
    dst[((size_t)b * BQ_N + n0 + nl) * BQ_C + tx] = tile[tx][nl]; // coalesced over c
  }
}

// ---------------------------------------------------------------------------
// Ball query: first NS indices (ascending m) with d2 < R2; pad with first hit.
// Full P2[b] staged in LDS as float4 (64 KB) -> one ds_load_b128 per candidate.
// ---------------------------------------------------------------------------
__global__ __launch_bounds__(256) void ball_query_kernel(const float* __restrict__ P1,
                                                         const float* __restrict__ P2,
                                                         int* __restrict__ idxout) {
  __shared__ float4 sp[BQ_N];   // 64 KB
  int b = blockIdx.y;
  const float* p2b = P2 + (size_t)b * BQ_N * 3;
  for (int i = threadIdx.x; i < BQ_N; i += 256) {
    sp[i] = make_float4(p2b[i * 3 + 0], p2b[i * 3 + 1], p2b[i * 3 + 2], 0.0f);
  }
  __syncthreads();

  int n = blockIdx.x * 256 + threadIdx.x;
  int p = b * BQ_N + n;
  float qx = P1[(size_t)p * 3 + 0];
  float qy = P1[(size_t)p * 3 + 1];
  float qz = P1[(size_t)p * 3 + 2];

  int* op = idxout + (size_t)p * BQ_NS;
  int cnt = 0, first = 0;
  for (int m = 0; m < BQ_N; ++m) {
    float4 c = sp[m];                       // ds_load_b128, broadcast across lanes
    float dx = c.x - qx, dy = c.y - qy, dz = c.z - qz;
    float d2 = dx * dx + dy * dy + dz * dz;
    if (d2 < R2) {
      if (cnt == 0) first = m;
      op[cnt] = m;
      if (++cnt == BQ_NS) break;
    }
  }
  for (int j = cnt; j < BQ_NS; ++j) op[j] = first;   // first==0 when no hit
}

// ---------------------------------------------------------------------------
// Fused group + MLP + max-pool. One wave per query point; 8 waves per block.
// A (16x160 f16) built per-wave in LDS; packed W staged once per block in LDS.
// 5 K-steps x 4 N-tiles of v_wmma_f32_16x16x32_f16 per point.
// ---------------------------------------------------------------------------
__global__ __launch_bounds__(256) void fused_mlp(const float* __restrict__ P1,
                                                 const float* __restrict__ P2,
                                                 const float* __restrict__ bias,
                                                 const int* __restrict__ idx,
                                                 const float* __restrict__ S2T,
                                                 const float* __restrict__ X1T,
                                                 const _Float16* __restrict__ Wp,
                                                 float* __restrict__ out) {
  __shared__ alignas(16) _Float16 sA[WAVES][BQ_NS][KPAD];   // 40 KB
  __shared__ alignas(16) _Float16 sB[WP_ELEMS];             // 20 KB

  const int wave = threadIdx.x >> 5;
  const int lane = threadIdx.x & 31;
  const int row16 = lane & 15;
  const int half  = lane >> 4;

  const int p = blockIdx.x * WAVES + wave;   // grid = B*N/WAVES exactly
  const int b = p >> 12;                     // N = 4096
  const int n = p & (BQ_N - 1);

  // ---- stage packed W into LDS (whole block cooperates) ----
  {
    const uint4* wsrc = (const uint4*)Wp;
    uint4* wdst = (uint4*)sB;
    for (int i = threadIdx.x; i < WP_ELEMS * 2 / 16; i += 256) wdst[i] = wsrc[i];
  }

  // each lane keeps the gather index for row (lane&15)
  const int myidx = idx[(size_t)p * BQ_NS + row16];

  const float* s2b = S2T + (size_t)b * BQ_N * BQ_C;
  const float* x1p = X1T + (size_t)p * BQ_C;
  const float q0 = P1[(size_t)p * 3 + 0];
  const float q1 = P1[(size_t)p * 3 + 1];
  const float q2 = P1[(size_t)p * 3 + 2];

  // ---- build A = [S2_grouped | X1 | disp | 0-pad], f16, row-major in LDS ----
  for (int i = lane; i < BQ_NS * KPAD; i += 32) {
    int r = i / KPAD;
    int k = i - r * KPAD;
    int gi = __shfl(myidx, r, 32);
    float v;
    if (k < BQ_C) {
      v = s2b[(size_t)gi * BQ_C + k];
    } else if (k < 2 * BQ_C) {
      v = x1p[k - BQ_C];
    } else if (k < 2 * BQ_C + 3) {
      int d = k - 2 * BQ_C;
      float pc = P2[((size_t)b * BQ_N + gi) * 3 + d];
      float qc = (d == 0) ? q0 : (d == 1) ? q1 : q2;
      v = pc - qc;
    } else {
      v = 0.0f;
    }
    sA[wave][r][k] = (_Float16)v;
  }
  __syncthreads();

  // ---- bias-initialized accumulators (C value depends only on N = lane&15) ----
  v8f acc[4];
#pragma unroll
  for (int nt = 0; nt < 4; ++nt) {
    float bv = bias[nt * 16 + row16];
#pragma unroll
    for (int r = 0; r < 8; ++r) acc[nt][r] = bv;
  }

  // ---- K loop: 5 steps of 32, 4 N tiles -> 20 WMMAs per point ----
#pragma unroll
  for (int ks = 0; ks < 5; ++ks) {
    // A fragment: contiguous K per lane half (K = half*16+e), matching packed B
    const _Float16* ap = &sA[wave][row16][ks * 32 + half * 16];
    v8h alo = *(const v8h*)ap;          // ds_load_b128
    v8h ahi = *(const v8h*)(ap + 8);    // ds_load_b128
    v16h a = __builtin_shufflevector(alo, ahi, 0, 1, 2, 3, 4, 5, 6, 7,
                                     8, 9, 10, 11, 12, 13, 14, 15);
#pragma unroll
    for (int nt = 0; nt < 4; ++nt) {
      const _Float16* bp = sB + ((nt * 5 + ks) * 32 + lane) * 16;
      v8h blo = *(const v8h*)bp;        // ds_load_b128
      v8h bhi = *(const v8h*)(bp + 8);  // ds_load_b128
      v16h bb = __builtin_shufflevector(blo, bhi, 0, 1, 2, 3, 4, 5, 6, 7,
                                        8, 9, 10, 11, 12, 13, 14, 15);
      acc[nt] = __builtin_amdgcn_wmma_f32_16x16x32_f16(
          /*neg_a=*/false, a, /*neg_b=*/false, bb,
          /*c_mod=*/(short)0, acc[nt], /*reuse_a=*/false, /*reuse_b=*/false);
    }
  }

  // ---- ReLU + max over the 16 samples (M rows), write (B, OUT, N) ----
#pragma unroll
  for (int nt = 0; nt < 4; ++nt) {
    float m = acc[nt][0];
#pragma unroll
    for (int r = 1; r < 8; ++r) m = fmaxf(m, acc[nt][r]);      // rows M=r, M=r+8
    m = fmaxf(m, __shfl_xor(m, 16, 32));                       // join lane halves
    m = fmaxf(m, 0.0f);                                        // relu∘max = max∘relu
    if (lane < 16) {
      out[((size_t)b * BQ_OUT + nt * 16 + row16) * BQ_N + n] = m;
    }
  }
}

// ---------------------------------------------------------------------------
extern "C" void kernel_launch(void* const* d_in, const int* in_sizes, int n_in,
                              void* d_out, int out_size, void* d_ws, size_t ws_size,
                              hipStream_t stream) {
  const float* P1 = (const float*)d_in[0];   // (B,N,3)
  const float* P2 = (const float*)d_in[1];   // (B,N,3)
  const float* X1 = (const float*)d_in[2];   // (B,C,N)
  const float* S2 = (const float*)d_in[3];   // (B,C,N)
  const float* W  = (const float*)d_in[4];   // (OUT, 2C+3)
  const float* bv = (const float*)d_in[5];   // (OUT)
  float* out = (float*)d_out;                // (B,OUT,N)

  const size_t IDX_BYTES = (size_t)BQ_B * BQ_N * BQ_NS * sizeof(int);     // 2 MB
  const size_t FT_BYTES  = (size_t)BQ_B * BQ_N * BQ_C * sizeof(float);    // 8 MB
  char* ws = (char*)d_ws;
  int*      idxbuf = (int*)ws;
  float*    S2T    = (float*)(ws + IDX_BYTES);
  float*    X1T    = (float*)(ws + IDX_BYTES + FT_BYTES);
  _Float16* Wp     = (_Float16*)(ws + IDX_BYTES + 2 * FT_BYTES);

  // 1) pack weights into WMMA B layout (f16)
  pack_w<<<(WP_ELEMS + 255) / 256, 256, 0, stream>>>(W, Wp);

  // 2) transpose feature maps (B,C,N)->(B,N,C) for contiguous gathers
  dim3 trGrid(BQ_N / 64, BQ_B);
  transpose_cn<<<trGrid, 256, 0, stream>>>(S2, S2T);
  transpose_cn<<<trGrid, 256, 0, stream>>>(X1, X1T);

  // 3) ball query
  dim3 bqGrid(BQ_N / 256, BQ_B);
  ball_query_kernel<<<bqGrid, 256, 0, stream>>>(P1, P2, idxbuf);

  // 4) fused gather + MLP (WMMA) + max-pool
  fused_mlp<<<(BQ_B * BQ_N) / WAVES, 256, 0, stream>>>(P1, P2, bv, idxbuf,
                                                       S2T, X1T, Wp, out);
}
